// Attention_77111842832766
// MI455X (gfx1250) — matrix operand order repaired and checked by reading, not compiled, not run on hardware
//
#include <hip/hip_runtime.h>
#include <hip/hip_bf16.h>

// ---------------------------------------------------------------------------
// CDNA5 (gfx1250) attention: bf16 WMMA + TDM (tensor_load_to_lds) staging.
// Shapes fixed by the reference: B=2, S=2048, HIDDEN=2048, H=32, KV=4, D=64.
// ---------------------------------------------------------------------------

#define XSTR(x) STR(x)
#define STR(x) #x
#pragma message("CDNA5 probe: __clang_major__=" XSTR(__clang_major__))
#if __has_builtin(__builtin_amdgcn_tensor_load_to_lds)
#pragma message("CDNA5 probe: tensor_load_to_lds AVAILABLE")
#else
#pragma message("CDNA5 probe: tensor_load_to_lds MISSING")
#endif
#if __has_builtin(__builtin_amdgcn_s_wait_tensorcnt)
#pragma message("CDNA5 probe: s_wait_tensorcnt AVAILABLE")
#else
#pragma message("CDNA5 probe: s_wait_tensorcnt MISSING")
#endif
#if __has_builtin(__builtin_amdgcn_permlane16)
#pragma message("CDNA5 probe: permlane16 AVAILABLE")
#else
#pragma message("CDNA5 probe: permlane16 MISSING")
#endif

typedef __attribute__((ext_vector_type(16))) __bf16 v16bf;
typedef __attribute__((ext_vector_type(8)))  float  v8f;

#if defined(__gfx1250__) && __has_builtin(__builtin_amdgcn_tensor_load_to_lds)
#define CDNA5_USE_TDM 1
#else
#define CDNA5_USE_TDM 0
#endif

union BF16x16 {
    v16bf v;
    unsigned int d[8];
    unsigned short u[16];
};

__device__ inline unsigned short f2bf(float f) {
    unsigned int x = __float_as_uint(f);
    unsigned int r = x + 0x7FFFu + ((x >> 16) & 1u);   // round-to-nearest-even
    return (unsigned short)(r >> 16);
}

__device__ inline v8f wmma_bf16(v16bf a, v16bf b, v8f c) {
    // (neg_a, A, neg_b, B, c_mod, C, reuse_a, reuse_b)
    return __builtin_amdgcn_wmma_f32_16x16x32_bf16(false, a, false, b,
                                                   (short)0, c, false, false);
}

// Row-wide (16-lane group) max reduction on the VALU via v_permlane16_b32
// (xor-1/2/4/8 selector patterns); frees the LDS pipe for WMMA operand reads.
#if __has_builtin(__builtin_amdgcn_permlane16)
__device__ inline float permxor(float v, unsigned s0, unsigned s1) {
    return __uint_as_float(__builtin_amdgcn_permlane16(
        0u, __float_as_uint(v), s0, s1, false, false));
}
__device__ inline float rowmax16(float x) {
    x = fmaxf(x, permxor(x, 0x67452301u, 0xEFCDAB89u));  // ^1
    x = fmaxf(x, permxor(x, 0x54761032u, 0xDCFE98BAu));  // ^2
    x = fmaxf(x, permxor(x, 0x32107654u, 0xBA98FEDCu));  // ^4
    x = fmaxf(x, permxor(x, 0xFEDCBA98u, 0x76543210u));  // ^8
    return x;
}
#else
__device__ inline float rowmax16(float x) {
#pragma unroll
    for (int off = 8; off > 0; off >>= 1)
        x = fmaxf(x, __shfl_xor(x, off, 32));
    return x;
}
#endif

// A fragment (16x32 bf16) from row-major [m][k] storage, element stride given.
// ISA 7.12.2: lane m = lane&15; K pairs dword-packed -> merges to ds_load_b128.
__device__ inline v16bf frag_a_rowmajor(const unsigned short* smem, int stride,
                                        int row, int kbase, int lane) {
    const int m = lane & 15;
    const int half8 = (lane >> 4) << 3;
    const unsigned short* p = smem + (size_t)(row + m) * stride + kbase;
    BF16x16 u;
#pragma unroll
    for (int i = 0; i < 8; ++i) {
        const int kp = ((lane >> 4) << 4) + (i << 1);
        u.d[i] = *reinterpret_cast<const unsigned int*>(p + kp);
    }
    return u.v;
}

// B fragment (32x16 bf16) from [n][k] storage (k contiguous per row).
__device__ inline v16bf frag_b_kcontig(const unsigned short* smem, int stride,
                                       int col, int kbase, int lane) {
    const int n = lane & 15;
    const int kb = kbase + ((lane >> 4) << 4);
    const unsigned short* p = smem + (size_t)(col + n) * stride + kb;
    BF16x16 u;
#pragma unroll
    for (int i = 0; i < 8; ++i)
        u.d[i] = *reinterpret_cast<const unsigned int*>(p + 2 * i);
    return u.v;
}

// All-ones bf16 B fragment (for WMMA row sums: P @ ones -> rowsum tile).
__device__ inline v16bf frag_ones() {
    BF16x16 u;
#pragma unroll
    for (int i = 0; i < 8; ++i) u.d[i] = 0x3F803F80u;   // (1.0bf16, 1.0bf16)
    return u.v;
}

// ---------------------------------------------------------------------------
// Tensor Data Mover: 2-D tile load (rowlen=64 bf16 elems per row, LDS rows
// padded to 80 elems via pad_interval=32dw/pad_amount=8dw). D# packing per
// ISA 8.3/8.4; groups 2/3 zero (2-D tile). Handles both builtin arities.
// ---------------------------------------------------------------------------
#if CDNA5_USE_TDM
typedef __attribute__((ext_vector_type(4))) unsigned int u32x4;
typedef __attribute__((ext_vector_type(8))) int i32x8;
typedef __attribute__((ext_vector_type(4))) int i32x4;

__device__ inline void tdm_load_2d(unsigned lds_off, const void* gsrc,
                                   unsigned rows, unsigned rowlen,
                                   unsigned stride_elems) {
    const unsigned long long ga = (unsigned long long)(size_t)gsrc;
    u32x4 g0;
    g0[0] = 1u;                                            // count=1 (user D#)
    g0[1] = lds_off;                                       // lds_addr bytes
    g0[2] = (unsigned)(ga & 0xFFFFFFFFu);                  // global_addr[31:0]
    g0[3] = (unsigned)((ga >> 32) & 0x01FFFFFFu) | (2u << 30);  // [56:32]|type=2
    i32x8 g1;
    g1[0] = (1 << 16)      // data_size = 2 bytes
          | (1 << 20)      // pad_enable
          | (4 << 22)      // pad_interval: 32 DWORDs (=64 bf16 elems)
          | (7 << 25);     // pad_amount: 8 DWORDs (=16 bf16 elems) -> pitch 80
    g1[1] = (int)((rowlen & 0xFFFFu) << 16);               // tensor_dim0 lo16
    g1[2] = (int)((rowlen >> 16) | ((rows & 0xFFFFu) << 16));  // dim0 hi|dim1 lo
    g1[3] = (int)((rows >> 16) | ((rowlen & 0xFFFFu) << 16));  // dim1 hi|tile_dim0
    g1[4] = (int)(rows & 0xFFFFu);                         // tile_dim1 (dim2=0)
    g1[5] = (int)stride_elems;                             // tensor_dim0_stride
    g1[6] = 0;
    g1[7] = 0;
    const i32x4 z4 = {0, 0, 0, 0};
#if __clang_major__ >= 23
    const i32x8 z8 = {0, 0, 0, 0, 0, 0, 0, 0};
    __builtin_amdgcn_tensor_load_to_lds(g0, g1, z4, z4, z8, 0);
#else
    __builtin_amdgcn_tensor_load_to_lds(g0, g1, z4, z4, 0);
#endif
}

__device__ inline void wait_tensorcnt0() {
#if __has_builtin(__builtin_amdgcn_s_wait_tensorcnt)
    __builtin_amdgcn_s_wait_tensorcnt(0);
#else
    asm volatile("s_wait_tensorcnt 0" ::: "memory");
#endif
}
#endif

// ---------------------------------------------------------------------------
// fp32 -> bf16 pack (same layout)
// ---------------------------------------------------------------------------
__global__ void f32_to_bf16_kernel(const float* __restrict__ in,
                                   unsigned short* __restrict__ out, size_t n) {
    size_t i = (size_t)blockIdx.x * blockDim.x + threadIdx.x;
    if (i < n) out[i] = f2bf(in[i]);
}

// fp32 [2048][N] -> bf16 [N][2048] (weights pre-transposed so GEMM B-frags are
// k-contiguous). K = 2048 always -> masks/shifts, no division.
template <int N>
__global__ void transpose_f32_to_bf16_kernel(const float* __restrict__ in,
                                             unsigned short* __restrict__ out,
                                             size_t total) {
    size_t idx = (size_t)blockIdx.x * blockDim.x + threadIdx.x;
    if (idx >= total) return;
    const int k = (int)(idx & 2047);
    const size_t n = idx >> 11;
    out[(n << 11) + k] = f2bf(in[(size_t)k * N + n]);
}

// ---------------------------------------------------------------------------
// GEMM: C[M,N] f32 = A[M,K] bf16 @ Bt[N,K] bf16 (Bt = B transposed).
// Block tile 128x64, K-stage 64, 8 waves, each wave 32x32 (2x2 WMMA tiles).
// Double-buffered TDM staging (fallback: uint4 copies). LDS pitch 80.
// ---------------------------------------------------------------------------
#define GT_M 128
#define GT_N 64
#define GT_K 64
#define LP   80   // padded LDS row pitch (elems), 160B -> keeps 16B alignment

template <int N, int K>
__global__ __launch_bounds__(256) void gemm_bf16nt_kernel(
    const unsigned short* __restrict__ A, const unsigned short* __restrict__ Bt,
    float* __restrict__ C) {
    __shared__ __align__(16) unsigned short As[2][GT_M * LP];  // [m][k]
    __shared__ __align__(16) unsigned short Bs[2][GT_N * LP];  // [n][k]

    const int tid  = threadIdx.x;
    const int lane = tid & 31;
    const int wave = tid >> 5;
    const int wm = wave >> 1;  // 0..3 (M)
    const int wn = wave & 1;   // 0..1 (N)
    const int row0 = blockIdx.y * GT_M;
    const int col0 = blockIdx.x * GT_N;

    const v8f zero8 = {0.f, 0.f, 0.f, 0.f, 0.f, 0.f, 0.f, 0.f};
    v8f acc[2][2];
    acc[0][0] = zero8; acc[0][1] = zero8; acc[1][0] = zero8; acc[1][1] = zero8;

    const int nst = K / GT_K;
#if CDNA5_USE_TDM
    if (wave == 0) {
        tdm_load_2d((unsigned)(size_t)&As[0][0], A + (size_t)row0 * K, GT_M, GT_K, K);
        tdm_load_2d((unsigned)(size_t)&Bs[0][0], Bt + (size_t)col0 * K, GT_N, GT_K, K);
    }
#endif
#pragma unroll 2
    for (int st = 0; st < nst; ++st) {
        const int cur = st & 1;
        const int kk = st * GT_K;
#if CDNA5_USE_TDM
        if (wave == 0) wait_tensorcnt0();
        __syncthreads();
        if (st + 1 < nst && wave == 0) {
            tdm_load_2d((unsigned)(size_t)&As[cur ^ 1][0],
                        A + (size_t)row0 * K + kk + GT_K, GT_M, GT_K, K);
            tdm_load_2d((unsigned)(size_t)&Bs[cur ^ 1][0],
                        Bt + (size_t)col0 * K + kk + GT_K, GT_N, GT_K, K);
        }
#else
        __syncthreads();   // previous stage's consumers done
#pragma unroll
        for (int i = 0; i < 4; ++i) {
            const int e = (tid + i * 256) * 8;
            const int r = e >> 6, c = e & 63;
            *reinterpret_cast<uint4*>(&As[cur][r * LP + c]) =
                *reinterpret_cast<const uint4*>(A + (size_t)(row0 + r) * K + kk + c);
        }
#pragma unroll
        for (int i = 0; i < 2; ++i) {
            const int e = (tid + i * 256) * 8;
            const int r = e >> 6, c = e & 63;
            *reinterpret_cast<uint4*>(&Bs[cur][r * LP + c]) =
                *reinterpret_cast<const uint4*>(Bt + (size_t)(col0 + r) * K + kk + c);
        }
        __syncthreads();
#endif
#pragma unroll
        for (int kc = 0; kc < 2; ++kc) {
            v16bf a0 = frag_a_rowmajor(&As[cur][0], LP, wm * 32 +  0, kc * 32, lane);
            v16bf a1 = frag_a_rowmajor(&As[cur][0], LP, wm * 32 + 16, kc * 32, lane);
            v16bf b0 = frag_b_kcontig(&Bs[cur][0], LP, wn * 32 +  0, kc * 32, lane);
            v16bf b1 = frag_b_kcontig(&Bs[cur][0], LP, wn * 32 + 16, kc * 32, lane);
            acc[0][0] = wmma_bf16(a0, b0, acc[0][0]);
            acc[0][1] = wmma_bf16(a0, b1, acc[0][1]);
            acc[1][0] = wmma_bf16(a1, b0, acc[1][0]);
            acc[1][1] = wmma_bf16(a1, b1, acc[1][1]);
        }
    }

    // Epilogue: C layout — lane n = lane&15 (+half selects m+8), vgpr v = row.
    const int n  = lane & 15;
    const int mh = (lane >> 4) << 3;
#pragma unroll
    for (int tm = 0; tm < 2; ++tm)
#pragma unroll
        for (int tn = 0; tn < 2; ++tn) {
            float* cp = C + (size_t)(row0 + wm * 32 + tm * 16 + mh) * N
                          + col0 + wn * 32 + tn * 16 + n;
#pragma unroll
            for (int v = 0; v < 8; ++v)
                cp[(size_t)v * N] = acc[tm][tn][v];
        }
}

// ---------------------------------------------------------------------------
// RoPE (optional) + relayout: f32 [B,S,H*64] -> bf16 [B,H,S,64] (DMAJOR=0)
//                                         or -> bf16 [B,H,64,S] (DMAJOR=1, V)
// qscale folds the 1/sqrt(64) attention scale into Q. S = 2048 fixed.
// ---------------------------------------------------------------------------
template <int H, int DOROPE, int DMAJOR>
__global__ void rope_pack_kernel(const float* __restrict__ src,
                                 const float* __restrict__ cosT,
                                 const float* __restrict__ sinT,
                                 unsigned short* __restrict__ dst,
                                 float qscale, size_t total) {
    constexpr int HSH = __builtin_ctz((unsigned)H);
    size_t idx = (size_t)blockIdx.x * blockDim.x + threadIdx.x;
    if (idx >= total) return;
    const int pr = (int)(idx & 31);
    size_t t = idx >> 5;
    const int hh = (int)(t & (H - 1)); t >>= HSH;
    const int s  = (int)(t & 2047);
    const int bb = (int)(t >> 11);

    const float* sp = src + ((((size_t)bb << 11) + s) * H + hh) * 64 + (pr << 1);
    float e = sp[0], o = sp[1];
    float oe = e, oo = o;
    if (DOROPE) {
        const float c  = cosT[(s << 5) + pr];
        const float sn = sinT[(s << 5) + pr];
        oe = e * c - o * sn;
        oo = e * sn + o * c;
    }
    oe *= qscale;
    oo *= qscale;
    if (DMAJOR) {
        unsigned short* dp =
            dst + (((((size_t)bb * H + hh) << 6) + (pr << 1)) << 11) + s;
        dp[0]    = f2bf(oe);
        dp[2048] = f2bf(oo);
    } else {
        unsigned short* dp =
            dst + (((((size_t)bb * H + hh) << 11) + s) << 6) + (pr << 1);
        dp[0] = f2bf(oe);
        dp[1] = f2bf(oo);
    }
}

// ---------------------------------------------------------------------------
// Flash attention (causal, GQA n_rep=8), bf16 WMMA, online softmax.
// Grid: (S/128, B*32). 8 waves x 16 q rows. 64-key K/V tiles double-buffered
// in LDS via TDM. Q: [B,32,S,64] (pre-scaled); K: [B,4,S,64]; V: [B,4,64,S].
// Row sums of P computed by WMMA against a ones fragment (no shuffle tree).
// ---------------------------------------------------------------------------
#define AT_QB 128
#define SEQ   2048

__global__ __launch_bounds__(256) void flash_attn_kernel(
    const unsigned short* __restrict__ Qb, const unsigned short* __restrict__ Kb,
    const unsigned short* __restrict__ Vt, unsigned short* __restrict__ Ob) {
    __shared__ __align__(16) unsigned short Ks[2][64 * LP];    // [key][d]
    __shared__ __align__(16) unsigned short Vs[2][64 * LP];    // [d][key]
    __shared__ __align__(16) unsigned short Ps[8 * 16 * 64];   // per-wave P [m][key]

    const int tid  = threadIdx.x;
    const int lane = tid & 31;
    const int wave = tid >> 5;
    const int n16  = lane & 15;
    const int mh   = (lane >> 4) << 3;

    const int h   = blockIdx.y & 31;
    const int b   = blockIdx.y >> 5;
    const int kvh = h >> 3;                 // n_rep = 8
    const int q0  = blockIdx.x * AT_QB;
    const int qw  = q0 + wave * 16;

    // Per-wave Q fragments (16 rows x 64 d -> two 16x32 A-frags), from global.
    const unsigned short* qbaseP = Qb + (((size_t)(b * 32 + h) << 11) + qw) * 64;
    v16bf qf[2];
#pragma unroll
    for (int kc = 0; kc < 2; ++kc)
        qf[kc] = frag_a_rowmajor(qbaseP, 64, 0, kc * 32, lane);
    const v16bf ones = frag_ones();

    const v8f zero8 = {0.f, 0.f, 0.f, 0.f, 0.f, 0.f, 0.f, 0.f};
    v8f o[4];
    o[0] = zero8; o[1] = zero8; o[2] = zero8; o[3] = zero8;
    float mrow[8], lrow[8];
#pragma unroll
    for (int v = 0; v < 8; ++v) { mrow[v] = -1e30f; lrow[v] = 0.f; }

    const unsigned short* kp0 = Kb + (((size_t)(b * 4 + kvh)) << 11) * 64;  // [key][d]
    const unsigned short* vp0 = Vt + (((size_t)(b * 4 + kvh)) << 6) * SEQ;  // [d][S]

    const int ktmax = (q0 + AT_QB - 1) >> 6;   // inclusive (causal)
#if CDNA5_USE_TDM
    if (wave == 0) {
        tdm_load_2d((unsigned)(size_t)&Ks[0][0], kp0, 64, 64, 64);
        tdm_load_2d((unsigned)(size_t)&Vs[0][0], vp0, 64, 64, SEQ);
    }
#endif
    for (int kt = 0; kt <= ktmax; ++kt) {
        const int cur = kt & 1;
#if CDNA5_USE_TDM
        if (wave == 0) wait_tensorcnt0();
        __syncthreads();
        if (kt < ktmax && wave == 0) {
            tdm_load_2d((unsigned)(size_t)&Ks[cur ^ 1][0],
                        kp0 + (((size_t)kt + 1) << 12), 64, 64, 64);
            tdm_load_2d((unsigned)(size_t)&Vs[cur ^ 1][0],
                        vp0 + (kt + 1) * 64, 64, 64, SEQ);
        }
#else
        __syncthreads();
#pragma unroll
        for (int i = 0; i < 2; ++i) {
            const int e = (tid + i * 256) * 8;
            const int r = e >> 6, c = e & 63;
            *reinterpret_cast<uint4*>(&Ks[cur][r * LP + c]) =
                *reinterpret_cast<const uint4*>(kp0 + ((size_t)kt << 12) + e);
            *reinterpret_cast<uint4*>(&Vs[cur][r * LP + c]) =
                *reinterpret_cast<const uint4*>(vp0 + ((size_t)r << 11) + kt * 64 + c);
        }
        if (kt < ktmax)
            __builtin_prefetch(kp0 + (((size_t)kt + 1) << 12), 0, 1);
        __syncthreads();
#endif

        // Wave-uniform causal tile classification: rows qw..qw+15.
        const bool any_valid = (kt * 64) <= (qw + 15);   // else fully masked
        if (any_valid) {
            // Scores: 16x64 = 4 N-tiles, each 2 K-chunks of 32 over d.
            v8f sc[4];
            sc[0] = zero8; sc[1] = zero8; sc[2] = zero8; sc[3] = zero8;
#pragma unroll
            for (int tn = 0; tn < 4; ++tn)
#pragma unroll
                for (int kc = 0; kc < 2; ++kc) {
                    v16bf kf = frag_b_kcontig(&Ks[cur][0], LP, tn * 16, kc * 32, lane);
                    sc[tn] = wmma_bf16(qf[kc], kf, sc[tn]);
                }

            // Mask only tiles straddling the diagonal (Q pre-scaled by 1/8).
            if (kt * 64 + 63 > qw) {
#pragma unroll
                for (int tn = 0; tn < 4; ++tn) {
                    const int key = kt * 64 + tn * 16 + n16;
#pragma unroll
                    for (int v = 0; v < 8; ++v)
                        sc[tn][v] = (key <= qw + mh + v) ? sc[tn][v] : -1e30f;
                }
            }

            // Online softmax: row max via v_permlane16 (VALU), exp -> Ps.
            float corr[8];
#pragma unroll
            for (int v = 0; v < 8; ++v) {
                float x = fmaxf(fmaxf(sc[0][v], sc[1][v]), fmaxf(sc[2][v], sc[3][v]));
                x = rowmax16(x);
                const float mn = fmaxf(mrow[v], x);
                corr[v] = __expf(mrow[v] - mn);
                mrow[v] = mn;
#pragma unroll
                for (int tn = 0; tn < 4; ++tn) {
                    const float p = __expf(sc[tn][v] - mn);
                    Ps[wave * 1024 + (mh + v) * 64 + tn * 16 + n16] = f2bf(p);
                }
#pragma unroll
                for (int tn = 0; tn < 4; ++tn)
                    o[tn][v] *= corr[v];
            }

            // Wave-local P stores must land before DS reads of the A-frags.
            asm volatile("s_wait_dscnt 0" ::: "memory");

            // O += P @ V; row sums of P via WMMA against ones (every column of
            // the result holds the row sum -> lands on (vgpr=row, lane) layout).
            const unsigned short* pw = &Ps[wave * 1024];
            v8f srow = zero8;
#pragma unroll
            for (int kc = 0; kc < 2; ++kc) {
                v16bf pf = frag_a_rowmajor(pw, 64, 0, kc * 32, lane);
                srow = wmma_bf16(pf, ones, srow);
#pragma unroll
                for (int tn = 0; tn < 4; ++tn) {
                    v16bf vf = frag_b_kcontig(&Vs[cur][0], LP, tn * 16, kc * 32, lane);
                    o[tn] = wmma_bf16(pf, vf, o[tn]);
                }
            }
#pragma unroll
            for (int v = 0; v < 8; ++v)
                lrow[v] = lrow[v] * corr[v] + srow[v];
        }
    }

    // Normalize and emit bf16 O into [B,S,2048] for the Wo GEMM.
    float inv[8];
#pragma unroll
    for (int v = 0; v < 8; ++v) inv[v] = 1.0f / lrow[v];
#pragma unroll
    for (int tn = 0; tn < 4; ++tn) {
        unsigned short* op =
            Ob + ((((size_t)b << 11) + qw + mh) << 11) + h * 64 + tn * 16 + n16;
#pragma unroll
        for (int v = 0; v < 8; ++v)
            op[(size_t)v << 11] = f2bf(o[tn][v] * inv[v]);
    }
}

// ---------------------------------------------------------------------------
// Launch
// ---------------------------------------------------------------------------
extern "C" void kernel_launch(void* const* d_in, const int* in_sizes, int n_in,
                              void* d_out, int out_size, void* d_ws, size_t ws_size,
                              hipStream_t stream) {
    (void)in_sizes; (void)n_in; (void)out_size; (void)ws_size;

    constexpr int Bc = 2, Sc = 2048, HID = 2048, NH = 32, NKV = 4, HD = 64;
    constexpr int M = Bc * Sc;               // 4096 token rows

    const float* x    = (const float*)d_in[0];
    const float* cosT = (const float*)d_in[1];
    const float* sinT = (const float*)d_in[2];
    // d_in[3] = additive mask (causality handled analytically), d_in[8] = pos=0.
    const float* Wq   = (const float*)d_in[4];
    const float* Wk   = (const float*)d_in[5];
    const float* Wv   = (const float*)d_in[6];
    const float* Wo   = (const float*)d_in[7];
    float* out = (float*)d_out;

    // Workspace carve-up (~110 MB total, 256B aligned chunks).
    unsigned char* ws = (unsigned char*)d_ws;
    size_t off = 0;
    auto take = [&](size_t bytes) -> void* {
        void* p = ws + off;
        off = (off + bytes + 255) & ~(size_t)255;
        return p;
    };
    unsigned short* xb  = (unsigned short*)take((size_t)M * HID * 2);
    unsigned short* wqt = (unsigned short*)take((size_t)HID * NH * HD * 2);   // [N][K]
    unsigned short* wkt = (unsigned short*)take((size_t)HID * NKV * HD * 2);
    unsigned short* wvt = (unsigned short*)take((size_t)HID * NKV * HD * 2);
    unsigned short* wot = (unsigned short*)take((size_t)NH * HD * HID * 2);
    float* qf32 = (float*)take((size_t)M * NH * HD * 4);
    float* kf32 = (float*)take((size_t)M * NKV * HD * 4);
    float* vf32 = (float*)take((size_t)M * NKV * HD * 4);
    unsigned short* qbf = (unsigned short*)take((size_t)M * NH * HD * 2);   // [B,H,S,64]
    unsigned short* kbf = (unsigned short*)take((size_t)M * NKV * HD * 2);  // [B,KV,S,64]
    unsigned short* vtb = (unsigned short*)take((size_t)M * NKV * HD * 2);  // [B,KV,64,S]
    unsigned short* obf = (unsigned short*)take((size_t)M * NH * HD * 2);   // [B,S,2048]

    {   // x -> bf16 (row-major)
        size_t n = (size_t)M * HID;
        f32_to_bf16_kernel<<<(unsigned)((n + 255) / 256), 256, 0, stream>>>(x, xb, n);
    }
    {   // weights -> bf16 transposed [N][K]
        size_t nq = (size_t)HID * NH * HD;
        transpose_f32_to_bf16_kernel<NH * HD><<<(unsigned)((nq + 255) / 256), 256, 0,
                                                stream>>>(Wq, wqt, nq);
        size_t nk = (size_t)HID * NKV * HD;
        transpose_f32_to_bf16_kernel<NKV * HD><<<(unsigned)((nk + 255) / 256), 256, 0,
                                                 stream>>>(Wk, wkt, nk);
        transpose_f32_to_bf16_kernel<NKV * HD><<<(unsigned)((nk + 255) / 256), 256, 0,
                                                 stream>>>(Wv, wvt, nk);
        transpose_f32_to_bf16_kernel<NH * HD><<<(unsigned)((nq + 255) / 256), 256, 0,
                                                stream>>>(Wo, wot, nq);
    }

    // Projections (bf16 WMMA, f32 out), B pre-transposed.
    gemm_bf16nt_kernel<NH * HD, HID><<<dim3((NH * HD) / GT_N, M / GT_M), 256, 0,
                                       stream>>>(xb, wqt, qf32);
    gemm_bf16nt_kernel<NKV * HD, HID><<<dim3((NKV * HD) / GT_N, M / GT_M), 256, 0,
                                        stream>>>(xb, wkt, kf32);
    gemm_bf16nt_kernel<NKV * HD, HID><<<dim3((NKV * HD) / GT_N, M / GT_M), 256, 0,
                                        stream>>>(xb, wvt, vf32);

    // RoPE + head-major relayout into bf16 (V stored d-major; Q pre-scaled).
    {
        size_t tq = (size_t)Bc * Sc * NH * 32;
        rope_pack_kernel<NH, 1, 0><<<(unsigned)((tq + 255) / 256), 256, 0, stream>>>(
            qf32, cosT, sinT, qbf, 0.125f, tq);
        size_t tk = (size_t)Bc * Sc * NKV * 32;
        rope_pack_kernel<NKV, 1, 0><<<(unsigned)((tk + 255) / 256), 256, 0, stream>>>(
            kf32, cosT, sinT, kbf, 1.0f, tk);
        rope_pack_kernel<NKV, 0, 1><<<(unsigned)((tk + 255) / 256), 256, 0, stream>>>(
            vf32, cosT, sinT, vtb, 1.0f, tk);
    }

    // Causal flash attention.
    flash_attn_kernel<<<dim3(Sc / AT_QB, Bc * NH), 256, 0, stream>>>(
        qbf, kbf, vtb, obf);

    // Output projection.
    gemm_bf16nt_kernel<HID, NH * HD><<<dim3(HID / GT_N, M / GT_M), 256, 0, stream>>>(
        obf, wot, out);
}
